// GRN_22445499089386
// MI455X (gfx1250) — compile-verified, matrix operand order checked by reading, not run on hardware
//
#include <hip/hip_runtime.h>
#include <math.h>

// Shapes from the reference
#define T_STEPS 32
#define NBATCH  20000
#define FDIM    128
#define HDIM    128
#define CDIM    10

typedef _Float16 v8h  __attribute__((ext_vector_type(8)));
typedef _Float16 v16h __attribute__((ext_vector_type(16)));
typedef float    v8f  __attribute__((ext_vector_type(8)));

#define WAVES 4
#define ROWS_PER_BLOCK (16 * WAVES)   // 64 batch rows per workgroup -> 313 blocks

// XOR-swizzle on 16B chunks inside a 128-half row: the 16 lanes of a fragment
// read rows 256B apart; swizzling makes them hit 16 distinct bank groups.
__device__ __forceinline__ int swz(int row, int col) {
  return row * 128 + ((((col >> 3) ^ (row & 15)) << 3) | (col & 7));
}
__device__ __forceinline__ int swz16(int row, int c0) {  // c0 multiple of 8
  return row * 128 + (((c0 >> 3) ^ (row & 15)) << 3);
}

// One lane's 16 halves of a 16-bit A-fragment from swizzled LDS.
// Elements 0..7 = K k0..k0+7, elements 8..15 = K k0+16..k0+23
// (k0 already includes the +8 for lanes 16..31).
__device__ __forceinline__ v16h ldfrag_lds(const _Float16* base, int row, int k0) {
  const v8h lo = *(const v8h*)(base + swz16(row, k0));
  const v8h hi = *(const v8h*)(base + swz16(row, k0 + 16));
  return __builtin_shufflevector(lo, hi, 0,1,2,3,4,5,6,7,8,9,10,11,12,13,14,15);
}

// One lane's B-fragment slice (X^T) straight from f32 global memory.
__device__ __forceinline__ v16h ldfrag_x(const float* __restrict__ rowptr, int k0) {
  const float4 a0 = *(const float4*)(rowptr + k0);
  const float4 a1 = *(const float4*)(rowptr + k0 + 4);
  const float4 a2 = *(const float4*)(rowptr + k0 + 16);
  const float4 a3 = *(const float4*)(rowptr + k0 + 20);
  v16h r;
  r[0]=(_Float16)a0.x;  r[1]=(_Float16)a0.y;  r[2]=(_Float16)a0.z;  r[3]=(_Float16)a0.w;
  r[4]=(_Float16)a1.x;  r[5]=(_Float16)a1.y;  r[6]=(_Float16)a1.z;  r[7]=(_Float16)a1.w;
  r[8]=(_Float16)a2.x;  r[9]=(_Float16)a2.y;  r[10]=(_Float16)a2.z; r[11]=(_Float16)a2.w;
  r[12]=(_Float16)a3.x; r[13]=(_Float16)a3.y; r[14]=(_Float16)a3.z; r[15]=(_Float16)a3.w;
  return r;
}

// D-tile pair (f32) -> B-fragment (f16) for the next K=32 slice, purely in-lane:
// elements 0..7 come from tile 2s, elements 8..15 from tile 2s+1.
__device__ __forceinline__ v16h pack_pair(v8f a, v8f b) {
  v16h r;
  #pragma unroll
  for (int e = 0; e < 8; ++e) { r[e] = (_Float16)a[e]; r[8 + e] = (_Float16)b[e]; }
  return r;
}

#define WMMA_F16(A, B, C) \
  __builtin_amdgcn_wmma_f32_16x16x32_f16(false, (A), false, (B), (short)0, (C), false, false)

__global__ __launch_bounds__(128)   // unconstrained regs: deep pipeline, no spills
void rnn_attn_fused(const float* __restrict__ X,
                    const float* __restrict__ W_ih, const float* __restrict__ W_hh,
                    const float* __restrict__ b_ih, const float* __restrict__ b_hh,
                    const float* __restrict__ ln_g, const float* __restrict__ ln_b,
                    const float* __restrict__ attnW, const float* __restrict__ attnB,
                    const float* __restrict__ denseW, const float* __restrict__ denseB,
                    float* __restrict__ out) {
  __shared__ _Float16 sWih[128 * 128];     // 32 KB f16 swizzled (A operand)
  __shared__ _Float16 sWhh[128 * 128];     // 32 KB
  __shared__ _Float16 sWcat[16 * 128];     // 4 KB: row0=attn, 1..10=dense, gamma folded
  __shared__ float    sBias[128];          // b_ih + b_hh
  __shared__ float    sCB[16], sWS[16];

  const int tid   = threadIdx.x;
  const int lane  = tid & 31;
  const int wv    = tid >> 5;
  const int c     = lane & 15;         // batch column within the wave tile
  const int hi16  = (lane >> 4) & 1;
  const int k0b   = hi16 << 3;         // K offset for this lane half
  const int rbase = hi16 << 3;         // D-layout row base for this lane half

  // ---------------- one-time init ----------------
  for (int i = tid; i < 128 * 128; i += 128) {
    const int r = i >> 7, col = i & 127;
    sWih[swz(r, col)] = (_Float16)W_ih[i];
    sWhh[swz(r, col)] = (_Float16)W_hh[i];
  }
  if (tid < 128) sBias[tid] = b_ih[tid] + b_hh[tid];
  if (tid < 16) {
    const int o = tid;
    float cb = 0.f, ws = 0.f;
    for (int k = 0; k < 128; ++k) {
      float wc = 0.f;
      if (o == 0)       wc = attnW[k];
      else if (o <= 10) wc = denseW[(o - 1) * 128 + k];
      const float wcg = wc * ln_g[k];          // fold LayerNorm gamma
      sWcat[swz(o, k)] = (_Float16)wcg;
      cb += wc * ln_b[k];                      // fold LayerNorm beta
      ws += wcg;                               // for the -mu correction term
    }
    if (o == 0)       cb += attnB[0];
    else if (o <= 10) cb += denseB[o - 1];
    sCB[o] = cb;
    sWS[o] = ws;
  }
  __syncthreads();

  // Per-lane head constants: this lane owns output rows o = rbase..rbase+7.
  float ws8[8], cb8[8];
  #pragma unroll
  for (int r = 0; r < 8; ++r) { ws8[r] = sWS[rbase + r]; cb8[r] = sCB[rbase + r]; }

  // Head A-fragments (16 x 128, invariant) kept in registers.
  v16h wcatA[4];
  #pragma unroll
  for (int s = 0; s < 4; ++s) wcatA[s] = ldfrag_lds(sWcat, c, 32 * s + k0b);

  const int n0 = blockIdx.x * ROWS_PER_BLOCK + wv * 16;
  int nrow = n0 + c;
  if (nrow > NBATCH - 1) nrow = NBATCH - 1;   // clamp tail; stores masked at end

  // ---------------- h0 = W_ih·X[0]^T + (b_ih+b_hh), no relu ----------------
  v16h hfrag[4];   // recurrent state as ready-to-use B fragments (f16, registers)
  {
    v8f acc[8] = {};
    const float* xrow = X + (size_t)nrow * FDIM;   // t = 0
    #pragma unroll
    for (int kk = 0; kk < 128; kk += 32) {
      const v16h xb = ldfrag_x(xrow, kk + k0b);
      #pragma unroll
      for (int j = 0; j < 8; ++j) {
        const v16h wa = ldfrag_lds(sWih, j * 16 + c, kk + k0b);
        acc[j] = WMMA_F16(wa, xb, acc[j]);
      }
    }
    #pragma unroll
    for (int j = 0; j < 8; ++j) {
      const float4 b0 = *(const float4*)(sBias + j * 16 + rbase);
      const float4 b1 = *(const float4*)(sBias + j * 16 + rbase + 4);
      acc[j][0] += b0.x; acc[j][1] += b0.y; acc[j][2] += b0.z; acc[j][3] += b0.w;
      acc[j][4] += b1.x; acc[j][5] += b1.y; acc[j][6] += b1.z; acc[j][7] += b1.w;
    }
    #pragma unroll
    for (int s = 0; s < 4; ++s) hfrag[s] = pack_pair(acc[2 * s], acc[2 * s + 1]);
  }

  // Online softmax state (per batch column; replicated across the lane pair).
  float m_run = -__builtin_inff(), Zr = 0.f;
  float accO[8] = {};

  // ---------------- fused time loop ----------------
  for (int t = 0; t < T_STEPS; ++t) {
    const float* xrow = X + ((size_t)t * NBATCH + nrow) * FDIM;
    if (t + 1 < T_STEPS) {   // prefetch next timestep's full X row (4 cachelines)
      const char* nx = (const char*)(X + ((size_t)(t + 1) * NBATCH + nrow) * FDIM);
      __builtin_prefetch(nx, 0, 0);
      __builtin_prefetch(nx + 128, 0, 0);
      __builtin_prefetch(nx + 256, 0, 0);
      __builtin_prefetch(nx + 384, 0, 0);
    }

    v8f acc[8] = {};
    // W_ih · X[t]^T
    #pragma unroll
    for (int kk = 0; kk < 128; kk += 32) {
      const v16h xb = ldfrag_x(xrow, kk + k0b);
      #pragma unroll
      for (int j = 0; j < 8; ++j) {
        const v16h wa = ldfrag_lds(sWih, j * 16 + c, kk + k0b);
        acc[j] = WMMA_F16(wa, xb, acc[j]);
      }
    }
    // + W_hh · h_prev^T   (h_prev is register-resident B fragments)
    #pragma unroll
    for (int s = 0; s < 4; ++s) {
      const int kk = 32 * s;
      #pragma unroll
      for (int j = 0; j < 8; ++j) {
        const v16h wa = ldfrag_lds(sWhh, j * 16 + c, kk + k0b);
        acc[j] = WMMA_F16(wa, hfrag[s], acc[j]);
      }
    }

    // bias + relu + LayerNorm stats (this lane owns 64 of the 128 hidden values
    // of batch column c; the other 64 live in the paired lane -> one shfl_xor).
    float s1 = 0.f, s2 = 0.f;
    #pragma unroll
    for (int j = 0; j < 8; ++j) {
      const float4 b0 = *(const float4*)(sBias + j * 16 + rbase);
      const float4 b1 = *(const float4*)(sBias + j * 16 + rbase + 4);
      float bb[8] = {b0.x, b0.y, b0.z, b0.w, b1.x, b1.y, b1.z, b1.w};
      #pragma unroll
      for (int r = 0; r < 8; ++r) {
        float v = acc[j][r] + bb[r];
        v = v > 0.f ? v : 0.f;
        acc[j][r] = v;
        s1 += v;
        s2 += v * v;
      }
    }
    s1 += __shfl_xor(s1, 16, 32);
    s2 += __shfl_xor(s2, 16, 32);
    const float mu   = s1 * (1.0f / 128.0f);
    const float var_ = s2 * (1.0f / 128.0f) - mu * mu;
    const float rstd = rsqrtf(var_ + 1e-5f);

    // New recurrent state: in-lane repack into B fragments (no LDS!).
    #pragma unroll
    for (int s = 0; s < 4; ++s) hfrag[s] = pack_pair(acc[2 * s], acc[2 * s + 1]);

    // Head: G[o, batch] = (gamma ⊙ Wcat) · h^T, all operands in registers.
    v8f G = {};
    #pragma unroll
    for (int s = 0; s < 4; ++s) G = WMMA_F16(wcatA[s], hfrag[s], G);

    // logit_o = rstd*(G_o - mu*Σ gamma·Wcat_o) + (Σ Wcat_o·beta + bias_o)
    float lg[8];
    #pragma unroll
    for (int r = 0; r < 8; ++r) lg[r] = rstd * (G[r] - mu * ws8[r]) + cb8[r];

    // attn logit (o == 0) lives in lane c (< 16); broadcast to the pair.
    const float sAtt  = __shfl(lg[0], c, 32);
    const float nm    = fmaxf(m_run, sAtt);
    const float scale = __expf(m_run - nm);
    const float p     = __expf(sAtt - nm);
    Zr    = Zr * scale + p;
    m_run = nm;
    #pragma unroll
    for (int r = 0; r < 8; ++r) accO[r] = accO[r] * scale + p * lg[r];
  }

  // ---------------- write (N, 10): lanes<16 hold o=1..7, lanes>=16 hold o=8..10
  const int n = n0 + c;
  if (n < NBATCH) {
    const float inv = 1.0f / Zr;
    if (!hi16) {
      #pragma unroll
      for (int r = 1; r < 8; ++r) out[n * CDIM + (r - 1)] = accO[r] * inv;
    } else {
      #pragma unroll
      for (int r = 0; r < 3; ++r) out[n * CDIM + (7 + r)] = accO[r] * inv;
    }
  }
}

extern "C" void kernel_launch(void* const* d_in, const int* in_sizes, int n_in,
                              void* d_out, int out_size, void* d_ws, size_t ws_size,
                              hipStream_t stream) {
  (void)in_sizes; (void)n_in; (void)d_ws; (void)ws_size; (void)out_size;
  const float* X      = (const float*)d_in[0];
  const float* W_ih   = (const float*)d_in[1];
  const float* W_hh   = (const float*)d_in[2];
  const float* b_ih   = (const float*)d_in[3];
  const float* b_hh   = (const float*)d_in[4];
  const float* ln_g   = (const float*)d_in[5];
  const float* ln_b   = (const float*)d_in[6];
  const float* attnW  = (const float*)d_in[7];
  const float* attnB  = (const float*)d_in[8];
  const float* denseW = (const float*)d_in[9];
  const float* denseB = (const float*)d_in[10];
  float* out = (float*)d_out;

  const int grid = (NBATCH + ROWS_PER_BLOCK - 1) / ROWS_PER_BLOCK;  // 313
  rnn_attn_fused<<<grid, 128, 0, stream>>>(X, W_ih, W_hh, b_ih, b_hh,
                                           ln_g, ln_b, attnW, attnB,
                                           denseW, denseB, out);
}
